// Head_49744311222993
// MI455X (gfx1250) — matrix operand order repaired and checked by reading, not compiled, not run on hardware
//
#include <hip/hip_runtime.h>

// ---------------------------------------------------------------------------
// Causal attention head (B=4, T=4096, C=1024, H=64, cut=2048) for gfx1250.
// bf16 WMMA everywhere (v_wmma_f32_16x16x32_bf16), fp32 accumulation/softmax.
// Native __bf16 casts so the backend emits hardware cvt ops instead of
// bit-twiddled rounding; global_prefetch for the next key block.
// ---------------------------------------------------------------------------

typedef __bf16        v16bf __attribute__((ext_vector_type(16)));
typedef float         v8f   __attribute__((ext_vector_type(8)));
typedef unsigned int  v4u   __attribute__((ext_vector_type(4)));

union Frag {                  // 32 bytes = 8 VGPRs, one WMMA A or B operand
  v16bf  v;
  __bf16 u[16];
  v4u    q[2];
};

#define WMMA_BF16(A, B, C) \
  __builtin_amdgcn_wmma_f32_16x16x32_bf16(false, (A), false, (B), (short)0, (C), false, false)

#define NB   4
#define NT   4096
#define NC   1024
#define NH   64
#define NCUT 2048

// ---------------------------------------------------------------------------
// Kernel 0: Wq/Wk/Wv (1024x64 fp32, row-major) -> Wt (3 x 64 x 1024 bf16),
// i.e. N-major so WMMA B-fragments become contiguous 32B loads.
// ---------------------------------------------------------------------------
__global__ void prep_w_kernel(const float* __restrict__ Wq,
                              const float* __restrict__ Wk,
                              const float* __restrict__ Wv,
                              __bf16* __restrict__ Wt) {
  int i = blockIdx.x * blockDim.x + threadIdx.x;
  if (i >= 3 * NH * NC) return;
  int m = i >> 16;            // which matrix
  int n = (i >> 10) & 63;     // output column (H)
  int k = i & 1023;           // reduction index (C)
  const float* W = (m == 0) ? Wq : (m == 1 ? Wk : Wv);
  Wt[i] = (__bf16)W[k * NH + n];
}

// ---------------------------------------------------------------------------
// Kernel 1: QKV projection. One wave owns 16 rows of x; 12 accumulator tiles
// (q/k/v x 4 N-tiles of 16), K=1024 in 32 steps of wmma 16x16x32 bf16.
// Q,K stored row-major bf16; V stored transposed Vt[b][h][t] bf16.
// ---------------------------------------------------------------------------
__global__ __launch_bounds__(256) void qkv_proj_kernel(
    const float* __restrict__ x, const __bf16* __restrict__ Wt,
    const float* __restrict__ bq, const float* __restrict__ bk,
    const float* __restrict__ bv,
    __bf16* __restrict__ Qbf, __bf16* __restrict__ Kbf,
    __bf16* __restrict__ Vt) {
  const int lane = threadIdx.x & 31;
  const int wave = threadIdx.x >> 5;
  const int hi   = lane >> 4;      // which 16-lane half
  const int lm   = lane & 15;
  const int rowBase = (blockIdx.x * 8 + wave) * 16;   // 128 blocks x 8 waves x 16 rows

  const v8f vzero = {0.f, 0.f, 0.f, 0.f, 0.f, 0.f, 0.f, 0.f};
  v8f acc[3][4];
#pragma unroll
  for (int m = 0; m < 3; ++m)
#pragma unroll
    for (int nt = 0; nt < 4; ++nt) acc[m][nt] = vzero;

  const float* xrow = x + (size_t)(rowBase + lm) * NC;  // A: lane = row M

  for (int ks = 0; ks < 32; ++ks) {
    const int c0 = ks * 32 + hi * 16;   // A layout: contiguous K c0..c0+15
    Frag a;
#pragma unroll
    for (int j = 0; j < 8; ++j) {
      a.u[j]     = (__bf16)xrow[c0 + j];
      a.u[8 + j] = (__bf16)xrow[c0 + 8 + j];
    }
#pragma unroll
    for (int m = 0; m < 3; ++m) {
#pragma unroll
      for (int nt = 0; nt < 4; ++nt) {
        // B layout: lane = column N, contiguous K at hi*16
        const __bf16* wr =
            Wt + (size_t)(m * NH + nt * 16 + lm) * NC + ks * 32 + hi * 16;
        Frag bb;
        bb.q[0] = ((const v4u*)wr)[0];
        bb.q[1] = ((const v4u*)wr)[1];
        acc[m][nt] = WMMA_BF16(a.v, bb.v, acc[m][nt]);
      }
    }
  }

  const float* biases[3] = {bq, bk, bv};
#pragma unroll
  for (int m = 0; m < 3; ++m)
#pragma unroll
    for (int nt = 0; nt < 4; ++nt) {
      float bval = biases[m][nt * 16 + lm];   // bias depends on column only
#pragma unroll
      for (int g = 0; g < 8; ++g) acc[m][nt][g] += bval;
    }

  // C/D layout: VGPR g holds row M = g + 8*hi, column N = lane%16 (+16*nt)
#pragma unroll
  for (int g = 0; g < 8; ++g) {
    int r = rowBase + g + 8 * hi;
    int bi = r >> 12;              // / 4096
    int t  = r & (NT - 1);
#pragma unroll
    for (int nt = 0; nt < 4; ++nt) {
      int n = nt * 16 + lm;
      Qbf[(size_t)r * NH + n] = (__bf16)acc[0][nt][g];
      Kbf[(size_t)r * NH + n] = (__bf16)acc[1][nt][g];
      Vt[((size_t)(bi * NH + n) << 12) + t] = (__bf16)acc[2][nt][g];
    }
  }
}

// ---------------------------------------------------------------------------
// Kernel 2: flash attention. One wave owns a 16-query tile; iterates 64-key
// blocks (causally clipped): 8 WMMA for S=QK^T, fp32 online softmax with
// half-wave shfl_xor row reductions, P->LDS->A-fragment, 8 WMMA for O+=P*V.
// Next key block's K/V tiles are prefetched (global_prefetch_b8).
// ---------------------------------------------------------------------------
__global__ __launch_bounds__(256) void attn_kernel(
    const __bf16* __restrict__ Qbf, const __bf16* __restrict__ Kbf,
    const __bf16* __restrict__ Vt, float* __restrict__ out) {
  __shared__ __bf16 Plds[8][16 * NH];   // per-wave 16x64 bf16 P tile

  const int lane = threadIdx.x & 31;
  const int wave = threadIdx.x >> 5;
  const int hi   = lane >> 4;
  const int lm   = lane & 15;

  const int tile = blockIdx.x * 8 + wave;   // 512 tiles = 4 batches * 128
  const int b    = tile >> 7;
  const int q0   = (tile & 127) * 16;       // query offset within cut

  // Q A-fragments for H chunks [0,32) and [32,64)
  const __bf16* qrow = Qbf + (size_t)(b * NT + (NT - NCUT) + q0 + lm) * NH;
  Frag aQ0, aQ1;
  aQ0.q[0] = *(const v4u*)(qrow + hi * 16);
  aQ0.q[1] = *(const v4u*)(qrow + 8 + hi * 16);
  aQ1.q[0] = *(const v4u*)(qrow + 32 + hi * 16);
  aQ1.q[1] = *(const v4u*)(qrow + 40 + hi * 16);

  const v8f vzero = {0.f, 0.f, 0.f, 0.f, 0.f, 0.f, 0.f, 0.f};
  v8f O[4];
#pragma unroll
  for (int ht = 0; ht < 4; ++ht) O[ht] = vzero;
  float mrow[8], lrow[8];
#pragma unroll
  for (int g = 0; g < 8; ++g) { mrow[g] = -3.0e38f; lrow[g] = 0.f; }

  const float scale = 0.03125f;                 // C^-0.5 = 1/32
  const int tq_hi = (NT - NCUT) + q0 + 15;      // abs pos of last query row
  const int nkb = (tq_hi + 64) >> 6;            // causally-needed 64-key blocks

  for (int kb = 0; kb < nkb; ++kb) {
    const int keybase = kb << 6;

    // ---- prefetch next key block's K and V cachelines ----
    if (kb + 1 < nkb) {
      const int nb = keybase + 64;
      // K: 64 rows x 128B -> each lane prefetches every other row (256B lines)
      __builtin_prefetch(Kbf + (size_t)(b * NT + nb + lane * 2) * NH, 0, 1);
      // V: 64 h-rows, 128B span each -> two h-rows per lane
      __builtin_prefetch(Vt + ((size_t)(b * NH + lane) << 12) + nb, 0, 1);
      __builtin_prefetch(Vt + ((size_t)(b * NH + 32 + lane) << 12) + nb, 0, 1);
    }

    // ---- S = Q K^T over this 64-key block (4 subtiles of 16 keys) ----
    v8f S[4];
#pragma unroll
    for (int sub = 0; sub < 4; ++sub) {
      const __bf16* krow =
          Kbf + (size_t)(b * NT + keybase + sub * 16 + lm) * NH;  // lane = key col
      Frag bk0, bk1;
      const v4u* p0 = (const v4u*)(krow + hi * 16);        // H chunk 0
      const v4u* p1 = (const v4u*)(krow + 32 + hi * 16);   // H chunk 1
      bk0.q[0] = p0[0]; bk0.q[1] = p0[1];
      bk1.q[0] = p1[0]; bk1.q[1] = p1[1];
      v8f s = vzero;
      s = WMMA_BF16(aQ0.v, bk0.v, s);
      s = WMMA_BF16(aQ1.v, bk1.v, s);
      S[sub] = s;
    }

    // ---- scale + causal mask ----
#pragma unroll
    for (int sub = 0; sub < 4; ++sub) {
      int kp = keybase + sub * 16 + lm;
#pragma unroll
      for (int g = 0; g < 8; ++g) {
        int tq = (NT - NCUT) + q0 + g + 8 * hi;
        float v = S[sub][g] * scale;
        S[sub][g] = (kp > tq) ? -3.0e38f : v;
      }
    }

    // ---- online softmax (rows live in VGPR index g across a 16-lane half) --
#pragma unroll
    for (int g = 0; g < 8; ++g) {
      float mx = fmaxf(fmaxf(S[0][g], S[1][g]), fmaxf(S[2][g], S[3][g]));
#pragma unroll
      for (int d = 1; d < 16; d <<= 1) mx = fmaxf(mx, __shfl_xor(mx, d, 32));
      float mnew  = fmaxf(mrow[g], mx);
      float alpha = __expf(mrow[g] - mnew);
      mrow[g] = mnew;
      float ps = 0.f;
#pragma unroll
      for (int sub = 0; sub < 4; ++sub) {
        float p = __expf(S[sub][g] - mnew);
        S[sub][g] = p;
        ps += p;
      }
#pragma unroll
      for (int d = 1; d < 16; d <<= 1) ps += __shfl_xor(ps, d, 32);
      lrow[g] = lrow[g] * alpha + ps;
#pragma unroll
      for (int ht = 0; ht < 4; ++ht) O[ht][g] *= alpha;
    }

    // ---- C-layout P -> LDS -> A-layout fragments ----
    __bf16* pl = &Plds[wave][0];
#pragma unroll
    for (int g = 0; g < 8; ++g)
#pragma unroll
      for (int sub = 0; sub < 4; ++sub)
        pl[(g + 8 * hi) * NH + sub * 16 + lm] = (__bf16)S[sub][g];
    asm volatile("s_wait_dscnt 0x0" ::: "memory");  // same-wave DS RAW

    const __bf16* prow = pl + lm * NH;      // A: lane = query row
    Frag aP0, aP1;
    aP0.q[0] = *(const v4u*)(prow + hi * 16);
    aP0.q[1] = *(const v4u*)(prow + 8 + hi * 16);
    aP1.q[0] = *(const v4u*)(prow + 32 + hi * 16);
    aP1.q[1] = *(const v4u*)(prow + 40 + hi * 16);

    // ---- O += P V  (V transposed: B-fragment loads are contiguous) ----
#pragma unroll
    for (int ht = 0; ht < 4; ++ht) {
      const __bf16* vrow =
          Vt + ((size_t)(b * NH + ht * 16 + lm) << 12) + keybase;  // lane = h col
      Frag bv0, bv1;
      const v4u* p0 = (const v4u*)(vrow + hi * 16);        // keys [0,32)
      const v4u* p1 = (const v4u*)(vrow + 32 + hi * 16);   // keys [32,64)
      bv0.q[0] = p0[0]; bv0.q[1] = p0[1];
      bv1.q[0] = p1[0]; bv1.q[1] = p1[1];
      O[ht] = WMMA_BF16(aP0.v, bv0.v, O[ht]);
      O[ht] = WMMA_BF16(aP1.v, bv1.v, O[ht]);
    }
  }

  // ---- normalize and store fp32 output [B][cut][H] ----
#pragma unroll
  for (int g = 0; g < 8; ++g) {
    float inv = 1.0f / lrow[g];
    float* orow = out + (size_t)(b * NCUT + q0 + g + 8 * hi) * NH;
#pragma unroll
    for (int ht = 0; ht < 4; ++ht) orow[ht * 16 + lm] = O[ht][g] * inv;
  }
}

// ---------------------------------------------------------------------------
extern "C" void kernel_launch(void* const* d_in, const int* in_sizes, int n_in,
                              void* d_out, int out_size, void* d_ws, size_t ws_size,
                              hipStream_t stream) {
  (void)in_sizes; (void)n_in; (void)out_size; (void)ws_size;
  const float* x  = (const float*)d_in[0];
  const float* Wq = (const float*)d_in[1];
  const float* bq = (const float*)d_in[2];
  const float* Wk = (const float*)d_in[3];
  const float* bk = (const float*)d_in[4];
  const float* Wv = (const float*)d_in[5];
  const float* bv = (const float*)d_in[6];
  float* out = (float*)d_out;

  // workspace layout (bf16): Wt[3*64*1024] | Q[16384*64] | K[16384*64] | Vt[4*64*4096]
  char* ws = (char*)d_ws;
  __bf16* Wt  = (__bf16*)(ws);
  __bf16* Qbf = (__bf16*)(ws + 393216);
  __bf16* Kbf = (__bf16*)(ws + 393216 + 2097152);
  __bf16* Vt  = (__bf16*)(ws + 393216 + 2 * 2097152);

  prep_w_kernel<<<dim3(768), dim3(256), 0, stream>>>(Wq, Wk, Wv, Wt);
  qkv_proj_kernel<<<dim3(128), dim3(256), 0, stream>>>(x, Wt, bq, bk, bv,
                                                       Qbf, Kbf, Vt);
  attn_kernel<<<dim3(64), dim3(256), 0, stream>>>(Qbf, Kbf, Vt, out);
}
